// OptimizedGraphAttentionLayer_81123342287406
// MI455X (gfx1250) — compile-verified, matrix operand order checked by reading, not run on hardware
//
#include <hip/hip_runtime.h>
#include <hip/hip_bf16.h>

// ---------------------------------------------------------------------------
// GAT layer for MI455X (gfx1250, wave32).
//   h|res = x @ [W | W_res]   -> bf16 WMMA (v_wmma_f32_16x16x32_bf16), f32 acc
//     A tile staged in LDS via GLOBAL_LOAD_ASYNC_TO_LDS_B128 (ASYNCcnt path)
//   attention logits, 3-pass segment softmax over edges, scatter-aggregate,
//   fused bias + residual + LayerNorm.
// ---------------------------------------------------------------------------

#define DIMV 256
#define HV 4
#define CV 64
#define NEG_SLOPE 0.2f
#define LN_EPS 1e-5f

typedef __attribute__((ext_vector_type(16))) __bf16 v16bf;
typedef __attribute__((ext_vector_type(8)))  float  v8f;

static __device__ __forceinline__ unsigned short f32_to_bf16(float f) {
  unsigned int u = __float_as_uint(f);
  u = (u + 0x7FFFu + ((u >> 16) & 1u)) >> 16;   // round-to-nearest-even
  return (unsigned short)u;
}

// monotonic float <-> uint mapping for atomicMax over signed floats
static __device__ __forceinline__ unsigned int ordf(float f) {
  unsigned int b = __float_as_uint(f);
  return (b & 0x80000000u) ? ~b : (b | 0x80000000u);
}
static __device__ __forceinline__ float unordf(unsigned int u) {
  unsigned int b = (u & 0x80000000u) ? (u & 0x7FFFFFFFu) : ~u;
  return __uint_as_float(b);
}

// generic pointer -> LDS byte offset (addrspacecast(3->0) = {aperture, offset})
static __device__ __forceinline__ unsigned int lds_off(const void* p) {
  return (unsigned int)(unsigned long long)(size_t)p;
}

static __device__ __forceinline__ void wait_asynccnt0() {
#if __has_builtin(__builtin_amdgcn_s_wait_asynccnt)
  __builtin_amdgcn_s_wait_asynccnt(0);
#else
  asm volatile("s_wait_asynccnt 0x0" ::: "memory");
#endif
}

// --- K0: build transposed concatenated weights in bf16: WT[j][k] -------------
__global__ void convert_w_kernel(const float* __restrict__ W,
                                 const float* __restrict__ W_res,
                                 unsigned short* __restrict__ WT) {
  int i = blockIdx.x * blockDim.x + threadIdx.x;      // 512*256 elements
  if (i >= 512 * 256) return;
  int j = i >> 8;          // output column 0..511
  int k = i & 255;         // K index
  float v = (j < 256) ? W[k * 256 + j] : W_res[k * 256 + (j - 256)];
  WT[(size_t)j * 256 + k] = f32_to_bf16(v);
}

// --- K1: x -> bf16 -----------------------------------------------------------
__global__ void convert_x_kernel(const float* __restrict__ x,
                                 unsigned short* __restrict__ xbf,
                                 long long n) {
  long long i = (long long)blockIdx.x * blockDim.x + threadIdx.x;
  if (i < n) xbf[i] = f32_to_bf16(x[i]);
}

// --- K2: hres[N,512] = xbf[N,256] @ WT^T  (bf16 WMMA, f32 accum) -------------
// Block = 256 threads (8 waves). Block tile: 16 rows x 128 cols; each wave
// owns one 16x16 tile. A tile (16x256 bf16 = 8 KB) staged in LDS directly via
// async global->LDS DMA (no VGPR roundtrip), shared by all 8 waves. B is read
// from the pre-transposed weight buffer (256 KB, L2-resident).
__global__ __launch_bounds__(256) void gemm_wmma_kernel(
    const unsigned short* __restrict__ xbf,
    const unsigned short* __restrict__ WT,
    float* __restrict__ hres, int Nrows) {
  __shared__ uint4 As[512];                 // 16 rows * 256 bf16
  const int rowBase = blockIdx.x * 16;
  const int lane = threadIdx.x & 31;
  const int wave = threadIdx.x >> 5;
  const int colBase = blockIdx.y * 128 + wave * 16;

  // cooperative A-tile stage: 512 uint4, 2 per thread (rows are contiguous)
  {
    const uint4* gA = (const uint4*)(xbf + (size_t)rowBase * 256);
    const int q0 = threadIdx.x, q1 = threadIdx.x + 256;
    if (rowBase + 16 <= Nrows) {
      // full tile: async global -> LDS, tracked by ASYNCcnt
      unsigned int l0 = lds_off(&As[q0]);
      unsigned int l1 = lds_off(&As[q1]);
      unsigned long long g0 = (unsigned long long)(size_t)(gA + q0);
      unsigned long long g1 = (unsigned long long)(size_t)(gA + q1);
      asm volatile(
          "global_load_async_to_lds_b128 %0, %1, off\n\t"
          "global_load_async_to_lds_b128 %2, %3, off"
          :: "v"(l0), "v"(g0), "v"(l1), "v"(g1)
          : "memory");
      wait_asynccnt0();
    } else {
      // ragged tail tile: synchronous path with zero fill
      uint4 z; z.x = z.y = z.z = z.w = 0u;
      As[q0] = (rowBase + (q0 >> 5) < Nrows) ? gA[q0] : z;
      As[q1] = (rowBase + (q1 >> 5) < Nrows) ? gA[q1] : z;
    }
  }
  __syncthreads();

  const int rc    = lane & 15;   // A row within tile / B column within tile
  const int khalf = lane >> 4;   // half-wave K split per ISA 16-bit A layout

  const char* lrow = (const char*)As + rc * 512;                  // A row bytes
  const unsigned short* bcol = WT + (size_t)(colBase + rc) * 256; // B column

  v8f acc = {0.f, 0.f, 0.f, 0.f, 0.f, 0.f, 0.f, 0.f};
#pragma unroll
  for (int kt = 0; kt < 8; ++kt) {          // K = 256, 32 per WMMA
    const int kb = kt * 32 + khalf * 8;     // lanes 0-15: K 0..7 & 16..23
    union { uint4 q[2]; v16bf v; } A, B;    // lanes 16-31: K 8..15 & 24..31
    A.q[0] = *(const uint4*)(lrow + (size_t)kb * 2);
    A.q[1] = *(const uint4*)(lrow + (size_t)(kb + 16) * 2);
    B.q[0] = *(const uint4*)(bcol + kb);
    B.q[1] = *(const uint4*)(bcol + kb + 16);
    acc = __builtin_amdgcn_wmma_f32_16x16x32_bf16(
        /*neg_a=*/false, A.v, /*neg_b=*/false, B.v,
        /*c_mod=*/(short)0, acc, /*reuse_a=*/false, /*reuse_b=*/false);
  }

  // C/D layout: lanes 0-15 -> M=r, N=lane; lanes 16-31 -> M=r+8, N=lane-16
#pragma unroll
  for (int r = 0; r < 8; ++r) {
    int m = rowBase + r + khalf * 8;
    if (m < Nrows) hres[(size_t)m * 512 + colBase + rc] = acc[r];
  }
}

// --- K3: per-node attention logits ------------------------------------------
__global__ void logits_kernel(const float* __restrict__ hres,
                              const float* __restrict__ a_src,
                              const float* __restrict__ a_dst,
                              float* __restrict__ alsrc,
                              float* __restrict__ aldst, int N) {
  int i = blockIdx.x * blockDim.x + threadIdx.x;   // over N*H
  if (i >= N * HV) return;
  int n = i >> 2, hd = i & 3;
  const float* hp = hres + (size_t)n * 512 + hd * CV;
  const float* as = a_src + hd * CV;
  const float* ad = a_dst + hd * CV;
  float s1 = 0.f, s2 = 0.f;
#pragma unroll 8
  for (int c = 0; c < CV; ++c) { float h = hp[c]; s1 += h * as[c]; s2 += h * ad[c]; }
  alsrc[i] = s1;
  aldst[i] = s2;
}

// --- K4: zero-init accumulators ---------------------------------------------
__global__ void init_kernel(float* __restrict__ out,
                            unsigned int* __restrict__ emaxU,
                            float* __restrict__ denom,
                            long long nOut, long long nNH) {
  long long i = (long long)blockIdx.x * blockDim.x + threadIdx.x;
  if (i < nOut) out[i] = 0.f;
  if (i < nNH) { emaxU[i] = 0u; denom[i] = 0.f; }   // ord-encoding min
}

// --- K5: edge pass 1: leaky-relu logits + segment max ------------------------
__global__ void edge_pass1_kernel(const long long* __restrict__ ei,
                                  long long E, long long EP,
                                  const float* __restrict__ alsrc,
                                  const float* __restrict__ aldst,
                                  float* __restrict__ ee,
                                  unsigned int* __restrict__ emaxU) {
  long long e = (long long)blockIdx.x * blockDim.x + threadIdx.x;
  if (e >= EP) return;
  long long s, d;
  if (e < E) { s = ei[e]; d = ei[E + e]; } else { s = d = e - E; }
#pragma unroll
  for (int hd = 0; hd < HV; ++hd) {
    float v = alsrc[s * HV + hd] + aldst[d * HV + hd];
    v = (v > 0.f) ? v : NEG_SLOPE * v;
    ee[e * HV + hd] = v;
    atomicMax(&emaxU[d * HV + hd], ordf(v));
  }
}

// --- K6: edge pass 2: exp + segment sum --------------------------------------
__global__ void edge_pass2_kernel(const long long* __restrict__ ei,
                                  long long E, long long EP,
                                  float* __restrict__ ee,
                                  const unsigned int* __restrict__ emaxU,
                                  float* __restrict__ denom) {
  long long e = (long long)blockIdx.x * blockDim.x + threadIdx.x;
  if (e >= EP) return;
  long long d = (e < E) ? ei[E + e] : e - E;
#pragma unroll
  for (int hd = 0; hd < HV; ++hd) {
    float ex = __expf(ee[e * HV + hd] - unordf(emaxU[d * HV + hd]));
    ee[e * HV + hd] = ex;
    atomicAdd(&denom[d * HV + hd], ex);
  }
}

// --- K7: edge pass 3: scatter-aggregate messages -----------------------------
// 64 threads per edge (one per channel), 4 edges per 256-thread block.
__global__ __launch_bounds__(256) void edge_pass3_kernel(
    const long long* __restrict__ ei, long long E, long long EP,
    const float* __restrict__ ee, const float* __restrict__ denom,
    const float* __restrict__ hres, float* __restrict__ out) {
  long long e = (long long)blockIdx.x * 4 + (threadIdx.x >> 6);
  int c = threadIdx.x & 63;
  if (e >= EP) return;
  long long s, d;
  if (e < E) { s = ei[e]; d = ei[E + e]; } else { s = d = e - E; }
#pragma unroll
  for (int hd = 0; hd < HV; ++hd) {
    float alpha = ee[e * HV + hd] / (denom[d * HV + hd] + 1e-16f);
    atomicAdd(&out[d * DIMV + hd * CV + c],
              alpha * hres[s * 512 + hd * CV + c]);
  }
}

// --- K8: fused bias + residual + LayerNorm (one block per row) ---------------
__global__ __launch_bounds__(256) void finalize_ln_kernel(
    float* __restrict__ out, const float* __restrict__ hres,
    const float* __restrict__ bias_gat, const float* __restrict__ b_res,
    const float* __restrict__ gamma, const float* __restrict__ beta) {
  int n = blockIdx.x, j = threadIdx.x;
  float y = out[(size_t)n * DIMV + j] + bias_gat[j] +
            hres[(size_t)n * 512 + 256 + j] + b_res[j];
  __shared__ float s1[256], s2[256];
  s1[j] = y; s2[j] = y * y;
  __syncthreads();
  for (int s = 128; s > 0; s >>= 1) {
    if (j < s) { s1[j] += s1[j + s]; s2[j] += s2[j + s]; }
    __syncthreads();
  }
  float mu = s1[0] * (1.0f / DIMV);
  float var = s2[0] * (1.0f / DIMV) - mu * mu;
  float inv = rsqrtf(var + LN_EPS);
  out[(size_t)n * DIMV + j] = gamma[j] * (y - mu) * inv + beta[j];
}

// ---------------------------------------------------------------------------
extern "C" void kernel_launch(void* const* d_in, const int* in_sizes, int n_in,
                              void* d_out, int out_size, void* d_ws, size_t ws_size,
                              hipStream_t stream) {
  const float*     x        = (const float*)d_in[0];
  const long long* ei       = (const long long*)d_in[1];   // int64 [2,E]
  const float*     W        = (const float*)d_in[2];
  const float*     a_src    = (const float*)d_in[3];
  const float*     a_dst    = (const float*)d_in[4];
  const float*     bias_gat = (const float*)d_in[5];
  const float*     W_res    = (const float*)d_in[6];
  const float*     b_res    = (const float*)d_in[7];
  const float*     gamma    = (const float*)d_in[8];
  const float*     beta     = (const float*)d_in[9];
  float* out = (float*)d_out;

  const int       N  = in_sizes[0] / DIMV;
  const long long E  = (long long)(in_sizes[1] / 2);
  const long long EP = E + N;
  (void)n_in; (void)out_size; (void)ws_size;

  char* w = (char*)d_ws;
  size_t off = 0;
  auto take = [&](size_t b) { size_t o = off; off += (b + 255) & ~(size_t)255; return o; };
  unsigned short* WT    = (unsigned short*)(w + take((size_t)512 * 256 * 2));
  unsigned short* xbf   = (unsigned short*)(w + take((size_t)N * DIMV * 2));
  float*          hres  = (float*)(w + take((size_t)N * 512 * 4));
  float*          alsrc = (float*)(w + take((size_t)N * HV * 4));
  float*          aldst = (float*)(w + take((size_t)N * HV * 4));
  unsigned int*   emaxU = (unsigned int*)(w + take((size_t)N * HV * 4));
  float*          denom = (float*)(w + take((size_t)N * HV * 4));
  float*          ee    = (float*)(w + take((size_t)EP * HV * 4));

  convert_w_kernel<<<(512 * 256 + 255) / 256, 256, 0, stream>>>(W, W_res, WT);
  convert_x_kernel<<<(unsigned)(((long long)N * DIMV + 255) / 256), 256, 0, stream>>>(
      x, xbf, (long long)N * DIMV);

  dim3 gg((N + 15) / 16, 4);
  gemm_wmma_kernel<<<gg, 256, 0, stream>>>(xbf, WT, hres, N);

  logits_kernel<<<(N * HV + 255) / 256, 256, 0, stream>>>(hres, a_src, a_dst,
                                                          alsrc, aldst, N);
  init_kernel<<<(unsigned)(((long long)N * DIMV + 255) / 256), 256, 0, stream>>>(
      out, emaxU, denom, (long long)N * DIMV, (long long)N * HV);

  edge_pass1_kernel<<<(unsigned)((EP + 255) / 256), 256, 0, stream>>>(
      ei, E, EP, alsrc, aldst, ee, emaxU);
  edge_pass2_kernel<<<(unsigned)((EP + 255) / 256), 256, 0, stream>>>(
      ei, E, EP, ee, emaxU, denom);
  edge_pass3_kernel<<<(unsigned)((EP + 3) / 4), 256, 0, stream>>>(
      ei, E, EP, ee, denom, hres, out);

  finalize_ln_kernel<<<N, 256, 0, stream>>>(out, hres, bias_gat, b_res, gamma, beta);
}
